// GAT4CGAN_43465069036209
// MI455X (gfx1250) — matrix operand order repaired and checked by reading, not compiled, not run on hardware
//
#include <hip/hip_runtime.h>
#include <hip/hip_bf16.h>
#include <math.h>

#define B_      64
#define M_      128
#define NFEAT   64
#define NHID    32
#define NHEADS  8
#define ENC     256
#define HALF_   63
#define KFC     32768   // M_*ENC
#define NFC     2048    // M_*NCLASS
#define ALPHA_  0.2f
#define KCHUNKS 16
#define KCH     2048    // KFC/KCHUNKS

typedef __attribute__((ext_vector_type(16))) _Float16 v16h;
typedef __attribute__((ext_vector_type(8)))  float    v8f;
typedef __attribute__((ext_vector_type(2)))  float    v2f;

__device__ __forceinline__ v8f wmma_f16(v16h a, v16h b, v8f c) {
  return __builtin_amdgcn_wmma_f32_16x16x32_f16(false, a, false, b, (short)0, c, false, false);
}
__device__ __forceinline__ v8f wmma_f32(v2f a, v2f b, v8f c) {
  return __builtin_amdgcn_wmma_f32_16x16x4_f32(false, a, false, b, (short)0, c, false, false);
}

union V16HU { v16h v; float4 f4[2]; };

// A-matrix 16x32 f16 fragment from row-major LDS (ld halves, 16B-aligned rows).
// Lane L: row=row0+L%16, g=L/16.  a[0..7] = halves @ K=k0+8g.. ; a[8..15] = @ K+16.
__device__ __forceinline__ v16h lds_load_a_f16(const _Float16* base, int row0, int ld, int k0) {
  const int lane = threadIdx.x & 31;
  const _Float16* p = base + (row0 + (lane & 15)) * ld + k0 + 8 * ((lane >> 4) & 1);
  V16HU u;
  u.f4[0] = *(const float4*)p;
  u.f4[1] = *(const float4*)(p + 16);
  return u.v;
}

// B-matrix 32x16 f16 fragment from TRANSPOSED ([n][k], k-contiguous) LDS.
// Lane L: col=col0+L%16, g=L/16; b[0..7] = halves @ K=k0+8g.. ; b[8..15] = @ K+16.
__device__ __forceinline__ v16h lds_load_bT_f16(const _Float16* baseT, int k0, int col0, int ldT) {
  const int lane = threadIdx.x & 31;
  const _Float16* p = baseT + (col0 + (lane & 15)) * ldT + k0 + 8 * ((lane >> 4) & 1);
  V16HU u;
  u.f4[0] = *(const float4*)p;
  u.f4[1] = *(const float4*)(p + 16);
  return u.v;
}

__device__ __forceinline__ uint4 pack_v8f_to_f16(v8f a) {
  union { _Float16 h[8]; uint4 u; } t;
#pragma unroll
  for (int j = 0; j < 8; ++j) t.h[j] = (_Float16)a[j];
  return t.u;
}

// ---------------------------------------------------------------------------
// Kernel 1: per-(b,head) GAT: Wh = x@W, attention softmax, h = elu(att@Wh)
// ---------------------------------------------------------------------------
#define XLD  72    // xs   [128][72]  (k-major rows)
#define WSLD 72    // WsT  [32][72]   (transposed: [n][k])
#define WTLD 136   // WhsT [32][136]  (transposed: [n][k])
#define ATLD 136   // atts [128][136]

__global__ __launch_bounds__(256) void gat_kernel(
    const float* __restrict__ x, const float* __restrict__ adj,
    const float* __restrict__ W_att, const float* __restrict__ a_att,
    float* __restrict__ h_enc)
{
  __shared__ __align__(16) _Float16 xs[M_ * XLD];
  __shared__ __align__(16) _Float16 WsT[NHID * WSLD];
  __shared__ __align__(16) _Float16 WhsT[NHID * WTLD];
  __shared__ __align__(16) _Float16 atts[M_ * ATLD];
  __shared__ float e_src[M_], e_dst[M_];
  __shared__ float redmax[M_][2], redsum[M_][2], rowsum[M_];

  const int b    = blockIdx.x >> 3;
  const int hh   = blockIdx.x & 7;
  const int tid  = threadIdx.x;
  const int lane = tid & 31;
  const int wave = tid >> 5;       // 8 waves

  // Stage x[b] (128x64) row-major f16; W_att[hh] (64x32) transposed f16
  const float* xb = x + (size_t)b * M_ * NFEAT;
  for (int i = tid; i < M_ * NFEAT; i += 256)
    xs[(i >> 6) * XLD + (i & 63)] = (_Float16)xb[i];
  const float* Wg = W_att + (size_t)hh * NFEAT * NHID;
  for (int i = tid; i < NFEAT * NHID; i += 256)
    WsT[(i & 31) * WSLD + (i >> 5)] = (_Float16)Wg[i];   // [n][k]
  __syncthreads();

  // Wh(128x32) = xs(128x64) @ Ws(64x32); wave w -> row tile w; store transposed
  {
    const int mt = wave;
    v8f acc0 = {}, acc1 = {};
#pragma unroll
    for (int kt = 0; kt < 2; ++kt) {
      v16h a  = lds_load_a_f16(xs, mt * 16, XLD, kt * 32);
      v16h b0 = lds_load_bT_f16(WsT, kt * 32, 0,  WSLD);
      v16h b1 = lds_load_bT_f16(WsT, kt * 32, 16, WSLD);
      acc0 = wmma_f16(a, b0, acc0);
      acc1 = wmma_f16(a, b1, acc1);
    }
    // C layout: lane holds n = lane%16 (acc0) / 16+n (acc1), rows m = mt*16+8g+j
    const int g = lane >> 4, nn = lane & 15;
    const int mcol = mt * 16 + 8 * g;
    *(uint4*)&WhsT[nn * WTLD + mcol]        = pack_v8f_to_f16(acc0);
    *(uint4*)&WhsT[(16 + nn) * WTLD + mcol] = pack_v8f_to_f16(acc1);
  }
  __syncthreads();

  // e_src/e_dst: one thread per node (Wh is transposed: WhsT[o][node])
  if (tid < M_) {
    const float* aa = a_att + (size_t)hh * 2 * NHID;
    float s = 0.f, d = 0.f;
#pragma unroll
    for (int o = 0; o < NHID; ++o) {
      const float w = (float)WhsT[o * WTLD + tid];
      s += w * aa[o];
      d += w * aa[NHID + o];
    }
    e_src[tid] = s; e_dst[tid] = d;
  }
  __syncthreads();

  // masked leaky-relu softmax; 2 threads/row, store exp(v - max) unnormalized
  {
    const int row = tid >> 1, hf = tid & 1;
    const float* arow = adj + ((size_t)b * M_ + row) * M_ + hf * 64;
    const float es = e_src[row];
    float mx = -3.0e38f;
    for (int m = 0; m < 64; ++m) {
      float v = es + e_dst[hf * 64 + m];
      v = v > 0.f ? v : ALPHA_ * v;
      v = arow[m] > 0.f ? v : -9.0e15f;
      mx = fmaxf(mx, v);
    }
    redmax[row][hf] = mx;
    __syncthreads();
    const float rmx = fmaxf(redmax[row][0], redmax[row][1]);
    float s = 0.f;
    for (int m = 0; m < 64; ++m) {
      float v = es + e_dst[hf * 64 + m];
      v = v > 0.f ? v : ALPHA_ * v;
      v = arow[m] > 0.f ? v : -9.0e15f;
      const float p = __expf(v - rmx);
      s += p;
      atts[row * ATLD + hf * 64 + m] = (_Float16)p;
    }
    redsum[row][hf] = s;
    __syncthreads();
    if (tid < M_) rowsum[tid] = redsum[tid][0] + redsum[tid][1];
  }
  __syncthreads();

  // h = elu((atts @ Wh) / rowsum); write to h_enc[b][node][hh*32+o]
  {
    const int mt = wave;
    v8f acc0 = {}, acc1 = {};
#pragma unroll
    for (int kt = 0; kt < 4; ++kt) {
      v16h a  = lds_load_a_f16(atts, mt * 16, ATLD, kt * 32);
      v16h b0 = lds_load_bT_f16(WhsT, kt * 32, 0,  WTLD);
      v16h b1 = lds_load_bT_f16(WhsT, kt * 32, 16, WTLD);
      acc0 = wmma_f16(a, b0, acc0);
      acc1 = wmma_f16(a, b1, acc1);
    }
    const int g = lane >> 4, nn = lane & 15;
#pragma unroll
    for (int j = 0; j < 8; ++j) {
      const int m = mt * 16 + j + 8 * g;
      const float inv = 1.f / rowsum[m];
      float v0 = acc0[j] * inv;
      float v1 = acc1[j] * inv;
      v0 = v0 > 0.f ? v0 : expm1f(v0);
      v1 = v1 > 0.f ? v1 : expm1f(v1);
      float* dst = h_enc + (((size_t)b * M_ + m) * ENC) + hh * NHID;
      dst[nn]      = v0;
      dst[16 + nn] = v1;
    }
  }
}

// ---------------------------------------------------------------------------
// Kernel 2: split-K fc with async-to-LDS double buffering.
// z_part[kc] = h(64 x 2048-chunk) @ fc_W(chunk x 64cols); fp32 WMMA 16x16x4.
// fc_W (256MB) streamed once with NT hint; h_enc stays hot in L2.
// ---------------------------------------------------------------------------
#define FLD 68
__global__ __launch_bounds__(256) void fc_kernel(
    const float* __restrict__ h_enc, const float* __restrict__ fc_W,
    float* __restrict__ z_part)
{
  __shared__ __align__(16) float As[2][64 * FLD];
  __shared__ __align__(16) float Bs[2][64 * FLD];
  const int tid = threadIdx.x, lane = tid & 31, wave = tid >> 5;
  const int n0 = blockIdx.x * 64;
  const int k0 = blockIdx.y * KCH;
  const int mt = wave & 3, ntb = (wave >> 2) * 2;

  const int g   = lane >> 4;
  const int rA  = mt * 16 + (lane & 15);
  const int nB0 = ntb * 16 + (lane & 15);

  // this thread's 4 slots of the 64x64 tile (1024 float4 slots / 256 threads)
  int slr[4], slc[4];
  unsigned ldsA[2][4], ldsB[2][4];
#pragma unroll
  for (int i = 0; i < 4; ++i) {
    const int e = tid + i * 256;
    slr[i] = e >> 4;             // row within tile (0..63)
    slc[i] = (e & 15) << 2;      // col within tile (multiple of 4)
#pragma unroll
    for (int bf = 0; bf < 2; ++bf) {
      ldsA[bf][i] = (unsigned)(size_t)&As[bf][slr[i] * FLD + slc[i]];
      ldsB[bf][i] = (unsigned)(size_t)&Bs[bf][slr[i] * FLD + slc[i]];
    }
  }

  // issue async loads of one 64x64 A tile + B tile into buffer bf
#define FC_ISSUE(bf, kb)                                                        \
  {                                                                             \
    _Pragma("unroll")                                                           \
    for (int i = 0; i < 4; ++i) {                                               \
      const int offA = (int)(((size_t)slr[i] * KFC + (kb) + slc[i]) * 4);       \
      asm volatile("global_load_async_to_lds_b128 %0, %1, %2"                   \
                   :: "v"(ldsA[bf][i]), "v"(offA), "s"(h_enc) : "memory");      \
      const int offB = (int)((((size_t)(kb) + slr[i]) * NFC + n0 + slc[i]) * 4);\
      asm volatile("global_load_async_to_lds_b128 %0, %1, %2 th:TH_LOAD_NT"     \
                   :: "v"(ldsB[bf][i]), "v"(offB), "s"(fc_W) : "memory");       \
    }                                                                           \
  }

  v8f acc0 = {}, acc1 = {};
  FC_ISSUE(0, k0)                                    // prologue: tile 0
  for (int kk = 0; kk < KCH / 64; ++kk) {
    const int cur = kk & 1;
    if (kk + 1 < KCH / 64) {
      FC_ISSUE(1 - cur, k0 + (kk + 1) * 64)          // prefetch next tile
      asm volatile("s_wait_asynccnt 8" ::: "memory");   // tile kk complete
    } else {
      asm volatile("s_wait_asynccnt 0" ::: "memory");
    }
    __syncthreads();                                 // LDS tile visible to all
    const float* A = As[cur];
    const float* Bt = Bs[cur];
#pragma unroll
    for (int ks = 0; ks < 16; ++ks) {
      const int kf = 4 * ks + 2 * g;
      v2f a;  a.x  = A[rA * FLD + kf];        a.y  = A[rA * FLD + kf + 1];
      v2f b0; b0.x = Bt[kf * FLD + nB0];      b0.y = Bt[(kf + 1) * FLD + nB0];
      v2f b1; b1.x = Bt[kf * FLD + nB0 + 16]; b1.y = Bt[(kf + 1) * FLD + nB0 + 16];
      acc0 = wmma_f32(a, b0, acc0);
      acc1 = wmma_f32(a, b1, acc1);
    }
    __syncthreads();                                 // done reading buf[cur]
  }
#undef FC_ISSUE

  const int nn = lane & 15;
#pragma unroll
  for (int j = 0; j < 8; ++j) {
    const int m = mt * 16 + j + 8 * g;
    const size_t base = ((size_t)blockIdx.y * 64 + m) * NFC + n0 + ntb * 16;
    z_part[base + nn]      = acc0[j];
    z_part[base + 16 + nn] = acc1[j];
  }
}

// Deterministic split-K reduction + bias
__global__ void z_reduce_kernel(const float* __restrict__ z_part,
                                const float* __restrict__ fc_b,
                                float* __restrict__ z)
{
  const int i = blockIdx.x * blockDim.x + threadIdx.x;
  if (i < B_ * NFC) {
    float s = fc_b[i & (NFC - 1)];
#pragma unroll
    for (int kc = 0; kc < KCHUNKS; ++kc) s += z_part[(size_t)kc * B_ * NFC + i];
    z[i] = s;
  }
}

// ---------------------------------------------------------------------------
// Kernel 3: zr(64x128) = z(64x2048) @ fcr_W(2048x128) + fcr_b  (z lives in L2)
// ---------------------------------------------------------------------------
__global__ __launch_bounds__(256) void zr_kernel(
    const float* __restrict__ z, const float* __restrict__ fcr_W,
    const float* __restrict__ fcr_b, float* __restrict__ zr)
{
  const int tid = threadIdx.x, lane = tid & 31, wave = tid >> 5;
  const int mt = wave & 3, nt = wave >> 2;
  const int g  = lane >> 4;
  const int nn = lane & 15;
  const int ncol = blockIdx.x * 32 + nt * 16 + nn;
  const int rA   = mt * 16 + nn;
  v8f acc = {};
  for (int k = 0; k < NFC; k += 4) {
    const int kf = k + 2 * g;
    v2f a; a.x = z[(size_t)rA * NFC + kf];        a.y = z[(size_t)rA * NFC + kf + 1];
    v2f b; b.x = fcr_W[(size_t)kf * M_ + ncol];   b.y = fcr_W[(size_t)(kf + 1) * M_ + ncol];
    acc = wmma_f32(a, b, acc);
  }
  const float bias = fcr_b[ncol];
#pragma unroll
  for (int j = 0; j < 8; ++j) {
    const int m = mt * 16 + j + 8 * g;
    zr[m * M_ + ncol] = acc[j] + bias;
  }
}

// ---------------------------------------------------------------------------
// Kernel 4: adj_dec (triu-gather product, built on the fly from zr in LDS)
//           @ ohe_W[h] + bias, fused log-softmax over class groups of 5.
//           Grid: 63 h-slices x 8 column-chunks of 80 (16 groups of 5).
// ---------------------------------------------------------------------------
#define ZLD 132
#define OLD 84
__global__ __launch_bounds__(128) void ohe_kernel(
    const float* __restrict__ zr, const float* __restrict__ ohe_W,
    const float* __restrict__ ohe_b, float* __restrict__ out)
{
  __shared__ __align__(16) float zrs[B_ * ZLD];
  __shared__ __align__(16) float os[B_ * OLD];
  const int hidx  = blockIdx.x;    // 0..62
  const int chunk = blockIdx.y;    // 0..7
  const int c0    = chunk * 80;
  const int tid = threadIdx.x, lane = tid & 31, wave = tid >> 5; // 4 waves

  for (int i = tid; i < B_ * M_; i += 128)
    zrs[(i >> 7) * ZLD + (i & 127)] = zr[i];
  __syncthreads();

  const float* Wb = ohe_W + (size_t)hidx * M_ * (M_ * 5);
  const int mt = wave;                       // batch-row tile
  const int bb = mt * 16 + (lane & 15);
  const int g  = lane >> 4;
  const int nn = lane & 15;

  v8f acc[5] = {};
  for (int ks = 0; ks < 32; ++ks) {
    const int k = 4 * ks + 2 * g;
    v2f a;
    {
      bool lo = (k <= hidx);
      int rI = lo ? (126 - hidx) : hidx;
      int cI = lo ? (127 - hidx + k) : k;
      a.x = zrs[bb * ZLD + rI] * zrs[bb * ZLD + cI];
      const int k1 = k + 1;
      lo = (k1 <= hidx);
      rI = lo ? (126 - hidx) : hidx;
      cI = lo ? (127 - hidx + k1) : k1;
      a.y = zrs[bb * ZLD + rI] * zrs[bb * ZLD + cI];
    }
#pragma unroll
    for (int nt = 0; nt < 5; ++nt) {
      const int gcol = c0 + nt * 16 + nn;
      v2f b; b.x = Wb[(size_t)k * 640 + gcol]; b.y = Wb[(size_t)(k + 1) * 640 + gcol];
      acc[nt] = wmma_f32(a, b, acc[nt]);
    }
  }
#pragma unroll
  for (int nt = 0; nt < 5; ++nt) {
    const float bias = ohe_b[hidx * 640 + c0 + nt * 16 + nn];
#pragma unroll
    for (int j = 0; j < 8; ++j) {
      const int m = mt * 16 + j + 8 * g;
      os[m * OLD + nt * 16 + nn] = acc[nt][j] + bias;
    }
  }
  __syncthreads();

  // log-softmax over each run of 5 classes; write final output
  for (int t = tid; t < B_ * 16; t += 128) {
    const int row = t >> 4, gi = t & 15;
    float v[5];
    float mx = -3.0e38f;
#pragma unroll
    for (int c = 0; c < 5; ++c) { v[c] = os[row * OLD + gi * 5 + c]; mx = fmaxf(mx, v[c]); }
    float s = 0.f;
#pragma unroll
    for (int c = 0; c < 5; ++c) s += __expf(v[c] - mx);
    const float lse = mx + __logf(s);
    const int node = chunk * 16 + gi;
    float* dst = out + (((size_t)row * HALF_ + hidx) * M_ + node) * 5;
#pragma unroll
    for (int c = 0; c < 5; ++c) dst[c] = v[c] - lse;
  }
}

// ---------------------------------------------------------------------------
extern "C" void kernel_launch(void* const* d_in, const int* in_sizes, int n_in,
                              void* d_out, int out_size, void* d_ws, size_t ws_size,
                              hipStream_t stream) {
  const float* x     = (const float*)d_in[0];
  const float* adj   = (const float*)d_in[1];
  const float* W_att = (const float*)d_in[2];
  const float* a_att = (const float*)d_in[3];
  const float* fc_W  = (const float*)d_in[4];
  const float* fc_b  = (const float*)d_in[5];
  const float* fcr_W = (const float*)d_in[6];
  const float* fcr_b = (const float*)d_in[7];
  const float* ohe_W = (const float*)d_in[8];
  const float* ohe_b = (const float*)d_in[9];
  float* out = (float*)d_out;

  float* ws     = (float*)d_ws;
  float* h_enc  = ws;                                   // 64*128*256   = 2,097,152 f
  float* z_part = h_enc + (size_t)B_ * M_ * ENC;        // 16*64*2048   = 2,097,152 f
  float* z      = z_part + (size_t)KCHUNKS * B_ * NFC;  // 64*2048      =   131,072 f
  float* zrbuf  = z + (size_t)B_ * NFC;                 // 64*128       =     8,192 f

  gat_kernel<<<dim3(B_ * NHEADS), dim3(256), 0, stream>>>(x, adj, W_att, a_att, h_enc);
  fc_kernel<<<dim3(NFC / 64, KCHUNKS), dim3(256), 0, stream>>>(h_enc, fc_W, z_part);
  z_reduce_kernel<<<dim3((B_ * NFC + 255) / 256), dim3(256), 0, stream>>>(z_part, fc_b, z);
  zr_kernel<<<dim3(4), dim3(256), 0, stream>>>(z, fcr_W, fcr_b, zrbuf);
  ohe_kernel<<<dim3(HALF_, 8), dim3(128), 0, stream>>>(zrbuf, ohe_W, ohe_b, out);
}
